// TFAttention_70755291234623
// MI455X (gfx1250) — matrix operand order repaired and checked by reading, not compiled, not run on hardware
//
#include <hip/hip_runtime.h>
#include <hip/hip_bf16.h>

// Problem constants (from reference)
#define B_   4
#define S_   2048
#define NX_  1024
#define H_   16
#define D_   64
#define P_   512
#define L_   (P_ + S_)     // 2560 total KV length

typedef __attribute__((ext_vector_type(16))) __bf16 v16bf;
typedef __attribute__((ext_vector_type(8)))  float  v8f;

// ---- helpers -------------------------------------------------------------

__device__ __forceinline__ unsigned short f2bf(float x) {
    union { float f; unsigned u; } v; v.f = x;
    unsigned r = v.u + 0x7FFFu + ((v.u >> 16) & 1u);   // round-to-nearest-even
    return (unsigned short)(r >> 16);
}

__device__ __forceinline__ void store_bf4(unsigned short* p, float4 v) {
    ushort4 u;
    u.x = f2bf(v.x); u.y = f2bf(v.y); u.z = f2bf(v.z); u.w = f2bf(v.w);
    *(ushort4*)p = u;
}

union FragU { uint4 u[2]; v16bf v; };

// A operand (16x32 bf16): lane m = lane%16, half = lane/16.
// elems 0..7  = row[half*8 .. +7], elems 8..15 = row[16 + half*8 .. +7].
// Caller passes (rowbase + half*8).
__device__ __forceinline__ v16bf frag_a(const unsigned short* p) {
    FragU f;
    f.u[0] = *(const uint4*)(p);
    f.u[1] = *(const uint4*)(p + 16);
    return f.v;
}

// B operand (32x16 bf16): lane n = lane%16, half = lane/16.
// elems 0..7 = K[half*8 .. +7], elems 8..15 = K[16 + half*8 .. +7]; caller passes (colbase + half*8).
__device__ __forceinline__ v16bf frag_b(const unsigned short* p) {
    FragU f;
    f.u[0] = *(const uint4*)(p);
    f.u[1] = *(const uint4*)(p + 16);
    return f.v;
}

#define WMMA_BF16(A, Bf, C, RA) \
    __builtin_amdgcn_wmma_f32_16x16x32_bf16(false, (A), false, (Bf), (short)0, (C), (RA), false)

// ---- kernel 1: QKV GEMM  qkv = x @ c_attn_w + b --------------------------
// M = B*S = 8192, K = 1024, N = 3072. Block tile 64x64, 4 waves x 16 rows.
// Double-buffered LDS, one barrier per K-step.
__global__ __launch_bounds__(128)
void qkv_gemm_kernel(const float* __restrict__ x, const float* __restrict__ w,
                     const float* __restrict__ bias,
                     float* __restrict__ ws_q, float* __restrict__ present) {
    __shared__ __align__(16) unsigned short As[2][64 * 32];   // [m][k]
    __shared__ __align__(16) unsigned short Bs[2][64 * 32];   // transposed [n][k]

    const int tid  = threadIdx.x;
    const int lane = tid & 31;
    const int wv   = tid >> 5;
    const int half = lane >> 4;
    const int ln   = lane & 15;
    const int n0   = blockIdx.x * 64;
    const int m0   = blockIdx.y * 64;

    v8f acc[4];
    #pragma unroll
    for (int nt = 0; nt < 4; ++nt)
        #pragma unroll
        for (int r = 0; r < 8; ++r) acc[nt][r] = 0.0f;

    auto load_tile = [&](int kk, int buf) {
        // A tile: 64x32 f32 -> bf16 (float4-vectorized)
        for (int i4 = tid; i4 < 64 * 8; i4 += 128) {
            int r = i4 >> 3, c4 = (i4 & 7) << 2;
            float4 vdat = *(const float4*)&x[(size_t)(m0 + r) * NX_ + kk + c4];
            store_bf4(&As[buf][r * 32 + c4], vdat);
        }
        // B tile: 32x64 of w[K][3N], stored transposed Bs[n][k]
        for (int i4 = tid; i4 < 32 * 16; i4 += 128) {
            int r = i4 >> 4, c4 = (i4 & 15) << 2;
            float4 vdat = *(const float4*)&w[(size_t)(kk + r) * (3 * NX_) + n0 + c4];
            Bs[buf][(c4 + 0) * 32 + r] = f2bf(vdat.x);
            Bs[buf][(c4 + 1) * 32 + r] = f2bf(vdat.y);
            Bs[buf][(c4 + 2) * 32 + r] = f2bf(vdat.z);
            Bs[buf][(c4 + 3) * 32 + r] = f2bf(vdat.w);
        }
    };

    const int nit = NX_ / 32;
    load_tile(0, 0);
    for (int it = 0; it < nit; ++it) {
        __syncthreads();
        if (it + 1 < nit) load_tile((it + 1) * 32, (it + 1) & 1);

        const unsigned short* a_ = As[it & 1];
        const unsigned short* b_ = Bs[it & 1];
        v16bf af = frag_a(&a_[(wv * 16 + ln) * 32 + half * 8]);
        v16bf bfr[4];
        #pragma unroll
        for (int nt = 0; nt < 4; ++nt)
            bfr[nt] = frag_b(&b_[(nt * 16 + ln) * 32 + half * 8]);
        acc[0] = WMMA_BF16(af, bfr[0], acc[0], false);
        acc[1] = WMMA_BF16(af, bfr[1], acc[1], true);   // reuse-A hint
        acc[2] = WMMA_BF16(af, bfr[2], acc[2], true);
        acc[3] = WMMA_BF16(af, bfr[3], acc[3], true);
    }

    // epilogue: route q / k / v
    #pragma unroll
    for (int nt = 0; nt < 4; ++nt) {
        #pragma unroll
        for (int r = 0; r < 8; ++r) {
            int row = m0 + wv * 16 + r + half * 8;       // global m: b*S + s
            int col = n0 + nt * 16 + ln;                 // in [0, 3072)
            float val = acc[nt][r] + bias[col];
            int b = row >> 11, s = row & (S_ - 1);
            int part = col >> 10, cc = col & (NX_ - 1);
            int h = cc >> 6, d = cc & 63;
            if (part == 0) {
                ws_q[((size_t)(b * H_ + h) * S_ + s) * D_ + d] = val;
            } else {
                float* dst = present + (size_t)(part - 1) * ((size_t)B_ * H_ * L_ * D_);
                dst[((size_t)(b * H_ + h) * L_ + (P_ + s)) * D_ + d] = val;
            }
        }
    }
}

// ---- kernel 2: copy past KV into present rows [0, P) ---------------------
__global__ void copy_past_kernel(const float4* __restrict__ past,
                                 float* __restrict__ present) {
    size_t idx = (size_t)blockIdx.x * blockDim.x + threadIdx.x;
    const size_t total4 = (size_t)2 * B_ * H_ * P_ * D_ / 4;
    if (idx >= total4) return;
    const size_t slice4 = (size_t)P_ * D_ / 4;       // float4 per (kv,b,h)
    size_t slice = idx / slice4;
    size_t rrem  = idx % slice4;
    float4 vdat = past[idx];
    float4* dst = (float4*)(present + slice * (size_t)L_ * D_);
    dst[rrem] = vdat;
}

// ---- kernel 3: flash attention per (b,h), 64-query tiles -----------------
__global__ __launch_bounds__(128)
void attn_kernel(const float* __restrict__ q,        // ws_q [B,H,S,D]
                 const float* __restrict__ present,  // [2,B,H,L,D]
                 float* __restrict__ attn_out) {     // ws_attn [B,S,NX] merged heads
    __shared__ __align__(16) unsigned short Qs[64 * 64];        // [q][d], pre-scaled
    __shared__ __align__(16) unsigned short Ks[2][64 * 64];     // [key][d]
    __shared__ __align__(16) unsigned short Vt[2][64 * 64];     // [d][key]
    __shared__ __align__(16) unsigned short Ps[4][16 * 64];     // per-wave probs [q][key]

    const int tid  = threadIdx.x;
    const int lane = tid & 31;
    const int wv   = tid >> 5;
    const int half = lane >> 4;
    const int ln   = lane & 15;
    const int q0   = blockIdx.x * 64;
    const int bh   = blockIdx.y;
    const int b    = bh >> 4, h = bh & 15;

    const float* qbase = q + (size_t)bh * S_ * D_ + (size_t)q0 * D_;
    const float* kbase = present + (size_t)bh * L_ * D_;
    const float* vbase = present + (size_t)(B_ * H_ + bh) * L_ * D_;

    auto load_kv = [&](int k0, int buf) {
        for (int i4 = tid; i4 < 64 * 16; i4 += 128) {
            int r = i4 >> 4, c4 = (i4 & 15) << 2;
            float4 kv = *(const float4*)&kbase[(size_t)(k0 + r) * D_ + c4];
            store_bf4(&Ks[buf][r * 64 + c4], kv);
            float4 vv = *(const float4*)&vbase[(size_t)(k0 + r) * D_ + c4];
            Vt[buf][(c4 + 0) * 64 + r] = f2bf(vv.x);
            Vt[buf][(c4 + 1) * 64 + r] = f2bf(vv.y);
            Vt[buf][(c4 + 2) * 64 + r] = f2bf(vv.z);
            Vt[buf][(c4 + 3) * 64 + r] = f2bf(vv.w);
        }
    };

    // stage scaled Q (scale = 1/sqrt(D) folded into q) and first K/V block
    for (int i4 = tid; i4 < 64 * 16; i4 += 128) {
        int r = i4 >> 4, c4 = (i4 & 15) << 2;
        float4 vdat = *(const float4*)&qbase[(size_t)r * D_ + c4];
        vdat.x *= 0.125f; vdat.y *= 0.125f; vdat.z *= 0.125f; vdat.w *= 0.125f;
        store_bf4(&Qs[r * 64 + c4], vdat);
    }
    load_kv(0, 0);
    __syncthreads();

    // Q fragments are loop-invariant: hoist out of the key-block loop
    v16bf qf0 = frag_a(&Qs[(wv * 16 + ln) * 64 + 0  + half * 8]);
    v16bf qf1 = frag_a(&Qs[(wv * 16 + ln) * 64 + 32 + half * 8]);

    float m_i[8], l_i[8];
    #pragma unroll
    for (int r = 0; r < 8; ++r) { m_i[r] = -3.0e38f; l_i[r] = 0.0f; }
    v8f o[4];
    #pragma unroll
    for (int nt = 0; nt < 4; ++nt)
        #pragma unroll
        for (int r = 0; r < 8; ++r) o[nt][r] = 0.0f;

    const int nkb = (P_ + q0) / 64 + 1;   // causal: keys j <= P + s
    for (int kb = 0; kb < nkb; ++kb) {
        const int k0 = kb * 64;
        if (kb > 0) __syncthreads();       // prev compute done before overwrite
        if (kb + 1 < nkb) load_kv((kb + 1) * 64, (kb + 1) & 1);
        const unsigned short* Ksb = Ks[kb & 1];
        const unsigned short* Vtb = Vt[kb & 1];

        // scores: 16 queries x 64 keys, grouped frag loads -> wmma bursts
        v8f sc[4];
        #pragma unroll
        for (int nt = 0; nt < 4; ++nt)
            #pragma unroll
            for (int r = 0; r < 8; ++r) sc[nt][r] = 0.0f;
        {
            v16bf bfr[4];
            #pragma unroll
            for (int nt = 0; nt < 4; ++nt)
                bfr[nt] = frag_b(&Ksb[(nt * 16 + ln) * 64 + 0 + half * 8]);
            sc[0] = WMMA_BF16(qf0, bfr[0], sc[0], false);
            sc[1] = WMMA_BF16(qf0, bfr[1], sc[1], true);
            sc[2] = WMMA_BF16(qf0, bfr[2], sc[2], true);
            sc[3] = WMMA_BF16(qf0, bfr[3], sc[3], true);
            #pragma unroll
            for (int nt = 0; nt < 4; ++nt)
                bfr[nt] = frag_b(&Ksb[(nt * 16 + ln) * 64 + 32 + half * 8]);
            sc[0] = WMMA_BF16(qf1, bfr[0], sc[0], false);
            sc[1] = WMMA_BF16(qf1, bfr[1], sc[1], true);
            sc[2] = WMMA_BF16(qf1, bfr[2], sc[2], true);
            sc[3] = WMMA_BF16(qf1, bfr[3], sc[3], true);
        }

        // causal mask + online softmax (row lives across the 16 lanes of a half)
        #pragma unroll
        for (int r = 0; r < 8; ++r) {
            int s_glob = q0 + wv * 16 + r + half * 8;
            float rmax = -3.0e38f;
            #pragma unroll
            for (int nt = 0; nt < 4; ++nt) {
                int j_glob = k0 + nt * 16 + ln;
                float v = sc[nt][r];
                if (j_glob > P_ + s_glob) v = -10000.0f;   // reference mask value
                sc[nt][r] = v;
                rmax = fmaxf(rmax, v);
            }
            #pragma unroll
            for (int dd = 1; dd < 16; dd <<= 1)
                rmax = fmaxf(rmax, __shfl_xor(rmax, dd, 32));
            float mnew  = fmaxf(m_i[r], rmax);
            float alpha = __expf(m_i[r] - mnew);
            m_i[r] = mnew;
            float psum = 0.0f;
            #pragma unroll
            for (int nt = 0; nt < 4; ++nt) {
                float p = __expf(sc[nt][r] - mnew);
                psum += p;
                Ps[wv][(r + half * 8) * 64 + nt * 16 + ln] = f2bf(p);
            }
            #pragma unroll
            for (int dd = 1; dd < 16; dd <<= 1)
                psum += __shfl_xor(psum, dd, 32);
            l_i[r] = l_i[r] * alpha + psum;
            #pragma unroll
            for (int nt = 0; nt < 4; ++nt) o[nt][r] *= alpha;
        }

        // o += P @ V   (wave-local LDS round-trip; same-wave dep, no barrier)
        v16bf pf0 = frag_a(&Ps[wv][ln * 64 + 0  + half * 8]);
        v16bf pf1 = frag_a(&Ps[wv][ln * 64 + 32 + half * 8]);
        {
            v16bf bfr[4];
            #pragma unroll
            for (int nt = 0; nt < 4; ++nt)
                bfr[nt] = frag_b(&Vtb[(nt * 16 + ln) * 64 + 0 + half * 8]);
            o[0] = WMMA_BF16(pf0, bfr[0], o[0], false);
            o[1] = WMMA_BF16(pf0, bfr[1], o[1], true);
            o[2] = WMMA_BF16(pf0, bfr[2], o[2], true);
            o[3] = WMMA_BF16(pf0, bfr[3], o[3], true);
            #pragma unroll
            for (int nt = 0; nt < 4; ++nt)
                bfr[nt] = frag_b(&Vtb[(nt * 16 + ln) * 64 + 32 + half * 8]);
            o[0] = WMMA_BF16(pf1, bfr[0], o[0], false);
            o[1] = WMMA_BF16(pf1, bfr[1], o[1], true);
            o[2] = WMMA_BF16(pf1, bfr[2], o[2], true);
            o[3] = WMMA_BF16(pf1, bfr[3], o[3], true);
        }
    }

    // normalize and write merged-head layout [B,S,NX]
    #pragma unroll
    for (int r = 0; r < 8; ++r) {
        int s_glob = q0 + wv * 16 + r + half * 8;
        float inv = 1.0f / l_i[r];
        #pragma unroll
        for (int nt = 0; nt < 4; ++nt)
            attn_out[((size_t)b * S_ + s_glob) * NX_ + h * 64 + nt * 16 + ln] =
                o[nt][r] * inv;
    }
}

// ---- kernel 4: output projection  a = attn @ c_proj_w + b ----------------
__global__ __launch_bounds__(128)
void proj_gemm_kernel(const float* __restrict__ x, const float* __restrict__ w,
                      const float* __restrict__ bias, float* __restrict__ out) {
    __shared__ __align__(16) unsigned short As[2][64 * 32];
    __shared__ __align__(16) unsigned short Bs[2][64 * 32];

    const int tid  = threadIdx.x;
    const int lane = tid & 31;
    const int wv   = tid >> 5;
    const int half = lane >> 4;
    const int ln   = lane & 15;
    const int n0   = blockIdx.x * 64;
    const int m0   = blockIdx.y * 64;

    v8f acc[4];
    #pragma unroll
    for (int nt = 0; nt < 4; ++nt)
        #pragma unroll
        for (int r = 0; r < 8; ++r) acc[nt][r] = 0.0f;

    auto load_tile = [&](int kk, int buf) {
        for (int i4 = tid; i4 < 64 * 8; i4 += 128) {
            int r = i4 >> 3, c4 = (i4 & 7) << 2;
            float4 vdat = *(const float4*)&x[(size_t)(m0 + r) * NX_ + kk + c4];
            store_bf4(&As[buf][r * 32 + c4], vdat);
        }
        for (int i4 = tid; i4 < 32 * 16; i4 += 128) {
            int r = i4 >> 4, c4 = (i4 & 15) << 2;
            float4 vdat = *(const float4*)&w[(size_t)(kk + r) * NX_ + n0 + c4];
            Bs[buf][(c4 + 0) * 32 + r] = f2bf(vdat.x);
            Bs[buf][(c4 + 1) * 32 + r] = f2bf(vdat.y);
            Bs[buf][(c4 + 2) * 32 + r] = f2bf(vdat.z);
            Bs[buf][(c4 + 3) * 32 + r] = f2bf(vdat.w);
        }
    };

    const int nit = NX_ / 32;
    load_tile(0, 0);
    for (int it = 0; it < nit; ++it) {
        __syncthreads();
        if (it + 1 < nit) load_tile((it + 1) * 32, (it + 1) & 1);

        const unsigned short* a_ = As[it & 1];
        const unsigned short* b_ = Bs[it & 1];
        v16bf af = frag_a(&a_[(wv * 16 + ln) * 32 + half * 8]);
        v16bf bfr[4];
        #pragma unroll
        for (int nt = 0; nt < 4; ++nt)
            bfr[nt] = frag_b(&b_[(nt * 16 + ln) * 32 + half * 8]);
        acc[0] = WMMA_BF16(af, bfr[0], acc[0], false);
        acc[1] = WMMA_BF16(af, bfr[1], acc[1], true);
        acc[2] = WMMA_BF16(af, bfr[2], acc[2], true);
        acc[3] = WMMA_BF16(af, bfr[3], acc[3], true);
    }

    #pragma unroll
    for (int nt = 0; nt < 4; ++nt) {
        #pragma unroll
        for (int r = 0; r < 8; ++r) {
            int row = m0 + wv * 16 + r + half * 8;
            int col = n0 + nt * 16 + ln;
            out[(size_t)row * NX_ + col] = acc[nt][r] + bias[col];
        }
    }
}

// ---- launcher ------------------------------------------------------------
extern "C" void kernel_launch(void* const* d_in, const int* in_sizes, int n_in,
                              void* d_out, int out_size, void* d_ws, size_t ws_size,
                              hipStream_t stream) {
    const float* x  = (const float*)d_in[0];   // [B,S,NX]
    const float* lp = (const float*)d_in[1];   // [2,B,H,P,D]
    const float* aw = (const float*)d_in[2];   // [NX, 3NX]
    const float* ab = (const float*)d_in[3];   // [3NX]
    const float* pw = (const float*)d_in[4];   // [NX, NX]
    const float* pb = (const float*)d_in[5];   // [NX]

    float* out = (float*)d_out;
    const size_t a_size = (size_t)B_ * S_ * NX_;          // 8,388,608
    float* present = out + a_size;                        // [2,B,H,L,D]

    float* ws_q    = (float*)d_ws;                        // [B,H,S,D]
    float* ws_attn = ws_q + (size_t)B_ * H_ * S_ * D_;    // [B,S,NX]

    // 1) QKV projection (writes ws_q and present rows [P, L))
    qkv_gemm_kernel<<<dim3(3 * NX_ / 64, (B_ * S_) / 64), dim3(128), 0, stream>>>(
        x, aw, ab, ws_q, present);

    // 2) past KV -> present rows [0, P)
    {
        const size_t total4 = (size_t)2 * B_ * H_ * P_ * D_ / 4;
        copy_past_kernel<<<(unsigned)((total4 + 255) / 256), 256, 0, stream>>>(
            (const float4*)lp, present);
    }

    // 3) causal flash attention -> ws_attn (merged heads)
    attn_kernel<<<dim3(S_ / 64, B_ * H_), dim3(128), 0, stream>>>(
        ws_q, present, ws_attn);

    // 4) output projection -> a
    proj_gemm_kernel<<<dim3(NX_ / 64, (B_ * S_) / 64), dim3(128), 0, stream>>>(
        ws_attn, pw, pb, out);
}
